// PerceiverARAttention_68736656605489
// MI455X (gfx1250) — compile-verified, matrix operand order checked by reading, not run on hardware
//
#include <hip/hip_runtime.h>

typedef _Float16 half_t;
typedef __attribute__((ext_vector_type(16))) _Float16 v16h;
typedef __attribute__((ext_vector_type(8))) float v8f;
typedef unsigned int u32x4 __attribute__((ext_vector_type(4)));
typedef int i32x4 __attribute__((ext_vector_type(4)));
typedef int i32x8 __attribute__((ext_vector_type(8)));

#define DIM 1024
#define HEADS 16
#define DH 64
#define SEQ_LEN 4096
#define LATENT 1024
#define HIST 3072
#define BATCH 2

#if defined(__gfx1250__) && __has_builtin(__builtin_amdgcn_tensor_load_to_lds) && \
    __has_builtin(__builtin_amdgcn_s_wait_tensorcnt)
#define USE_TDM 1
#else
#define USE_TDM 0
#endif

// D = A(16x32 f16) * B(32x16 f16) + C(16x16 f32)
__device__ __forceinline__ v8f wmma_f16(v16h a, v16h b, v8f c) {
  return __builtin_amdgcn_wmma_f32_16x16x32_f16(false, a, false, b, (short)0, c,
                                                false, false);
}

// 16x32 A-style fragment (also used for B with lane = N) from row-major f16.
// Lane L<16 -> row L, K chunks {0..7,16..23}; lane L+16 -> {8..15,24..31}.
__device__ __forceinline__ v16h frag_from_f16(const half_t* __restrict__ base,
                                              int stride, int kbase) {
  int lane = threadIdx.x & 31;
  int r = lane & 15;
  int cb = (lane & 16) ? 8 : 0;
  const half_t* p = base + (size_t)r * stride + kbase + cb;
  v16h out;
#pragma unroll
  for (int i = 0; i < 8; ++i) out[i] = p[i];
#pragma unroll
  for (int i = 0; i < 8; ++i) out[8 + i] = p[16 + i];
  return out;
}

#if USE_TDM
// Issue a 2D TDM tile load: nrows rows of row_len f16 elements (global row
// stride = row_stride elements) -> contiguous rows at LDS byte offset lds_off.
__device__ __forceinline__ void tdm_load_2d(unsigned lds_off,
                                            const half_t* gptr,
                                            unsigned row_len, unsigned nrows,
                                            unsigned row_stride) {
  unsigned long long ga = (unsigned long long)gptr;
  u32x4 g0;
  g0[0] = 1u;                                    // count=1, user descriptor
  g0[1] = lds_off;                               // D#.lds_addr (bytes)
  g0[2] = (unsigned)ga;                          // D#.global_addr[31:0]
  g0[3] = (unsigned)(ga >> 32) | (2u << 30);     // addr[56:32] | type=2
  i32x8 g1;
  g1[0] = 1 << 16;                               // data_size=1 (2 bytes)
  g1[1] = (int)((row_len & 0xFFFFu) << 16);      // tensor_dim0[15:0]
  g1[2] = (int)((row_len >> 16) | ((nrows & 0xFFFFu) << 16));  // |tensor_dim1
  g1[3] = (int)((nrows >> 16) | (row_len << 16));              // |tile_dim0
  g1[4] = (int)(nrows & 0xFFFFu);                // tile_dim1 (tile_dim2=0)
  g1[5] = (int)row_stride;                       // tensor_dim0_stride[31:0]
  g1[6] = 0;                                     // stride0[47:32]|stride1 lo
  g1[7] = 0;
  i32x4 z4 = {0, 0, 0, 0};
#if __clang_major__ >= 23
  i32x8 z8 = {0, 0, 0, 0, 0, 0, 0, 0};
  __builtin_amdgcn_tensor_load_to_lds(g0, g1, z4, z4, z8, 0);
#else
  __builtin_amdgcn_tensor_load_to_lds(g0, g1, z4, z4, 0);
#endif
}
#endif

// fp32 -> f16 streaming convert (one-shot; removes all cvt from GEMM loops).
__global__ __launch_bounds__(256) void cvt_kernel(const float* __restrict__ src,
                                                  half_t* __restrict__ dst,
                                                  int n) {
  int idx = (blockIdx.x * 256 + threadIdx.x) * 8;
  if (idx >= n) return;
#pragma unroll
  for (int i = 0; i < 8; ++i) dst[idx + i] = (half_t)src[idx + i];
}

// One wave computes a 16(rows) x 64(one head) tile of X @ W^T, applies RoPE
// and (optionally) LayerNorm, stores f16 row-major and (optionally) dim-major.
__global__ __launch_bounds__(32) void proj_kernel(
    const half_t* __restrict__ x16, const half_t* __restrict__ W16,
    const float* __restrict__ gamma, const float* __restrict__ beta,
    half_t* __restrict__ out_rm, half_t* __restrict__ out_cm,
    int seq_per_batch, int pos_base, int do_ln, float scale) {
  int h = blockIdx.x & (HEADS - 1);
  int tile = blockIdx.x >> 4;
  int row0 = tile * 16;
  int batch = row0 / seq_per_batch;
  int local = row0 - batch * seq_per_batch;
  int lane = threadIdx.x & 31;
  int col = lane & 15;
  int rh = (lane & 16) ? 8 : 0;

  const half_t* Ab = x16 + ((size_t)batch * SEQ_LEN + pos_base + local) * DIM;
  const half_t* Bb = W16 + (size_t)h * DH * DIM;

  v8f acc[4] = {};
  for (int k = 0; k < DIM; k += 32) {
    v16h a = frag_from_f16(Ab, DIM, k);
#pragma unroll
    for (int f = 0; f < 4; ++f) {
      v16h bf = frag_from_f16(Bb + (size_t)(f * 16) * DIM, DIM, k);
      acc[f] = wmma_f16(a, bf, acc[f]);
    }
  }

  float v[4][8];
#pragma unroll
  for (int f = 0; f < 4; ++f)
#pragma unroll
    for (int i = 0; i < 8; ++i) v[f][i] = acc[f][i];

  // RoPE: dim d pairs with d+32 (fragments f and f+2 of the same lane).
#pragma unroll
  for (int i = 0; i < 8; ++i) {
    float p = (float)(pos_base + local + rh + i);
#pragma unroll
    for (int f = 0; f < 2; ++f) {
      int d = f * 16 + col;  // d < 32
      float w = __powf(10000.0f, -(float)d * (1.0f / 32.0f));
      float s, c;
      __sincosf(p * w, &s, &c);
      float lo = v[f][i], hi = v[f + 2][i];
      v[f][i] = lo * c - hi * s;
      v[f + 2][i] = hi * c + lo * s;
    }
  }

  if (do_ln) {
#pragma unroll
    for (int i = 0; i < 8; ++i) {
      float sum = v[0][i] + v[1][i] + v[2][i] + v[3][i];
      float sq = v[0][i] * v[0][i] + v[1][i] * v[1][i] + v[2][i] * v[2][i] +
                 v[3][i] * v[3][i];
#pragma unroll
      for (int m = 1; m < 16; m <<= 1) {
        sum += __shfl_xor(sum, m, 32);
        sq += __shfl_xor(sq, m, 32);
      }
      float mean = sum * (1.0f / 64.0f);
      float var = sq * (1.0f / 64.0f) - mean * mean;
      float inv = rsqrtf(var + 1e-5f);
#pragma unroll
      for (int f = 0; f < 4; ++f) {
        int d = f * 16 + col;
        v[f][i] = (v[f][i] - mean) * inv * gamma[d] + beta[d];
      }
    }
  } else {
#pragma unroll
    for (int f = 0; f < 4; ++f)
#pragma unroll
      for (int i = 0; i < 8; ++i) v[f][i] *= scale;
  }

  size_t bh = (size_t)batch * HEADS + h;
#pragma unroll
  for (int i = 0; i < 8; ++i) {
    int pos = local + rh + i;
#pragma unroll
    for (int f = 0; f < 4; ++f) {
      int d = f * 16 + col;
      half_t hv = (half_t)v[f][i];
      out_rm[(bh * seq_per_batch + pos) * DH + d] = hv;
      if (out_cm)
        out_cm[(bh * DH + d) * (size_t)seq_per_batch + pos] = hv;
    }
  }
}

// Flash attention: 128-thread block = 4 waves = 4 query tiles of one (b,h).
// K/V blocks (32 keys) staged in LDS, shared by all 4 waves; TDM double-
// buffered DMA when available.
// LDS map (halves): kbuf[2][2048] @0 | vbuf[2][2048] @4096 | pbuf[4][512] @8192
__global__ __launch_bounds__(128) void attn_kernel(
    const half_t* __restrict__ q, const half_t* __restrict__ kvn,
    const half_t* __restrict__ kvnT, half_t* __restrict__ o) {
  __shared__ _Float16 smem[2 * 2048 + 2 * 2048 + 4 * 512];
  int qt4 = blockIdx.x & 15;  // group of 4 query tiles
  int bh = blockIdx.x >> 4;
  int wave = threadIdx.x >> 5;
  int lane = threadIdx.x & 31;
  int i0 = qt4 * 64 + wave * 16;
  int col = lane & 15;
  int rh = (lane & 16) ? 8 : 0;

  const half_t* qb = q + ((size_t)bh * LATENT + i0) * DH;
  const half_t* kb = kvn + (size_t)bh * SEQ_LEN * DH;
  const half_t* vb = kvnT + (size_t)bh * DH * SEQ_LEN;

  v16h qa0 = frag_from_f16(qb, DH, 0);
  v16h qa1 = frag_from_f16(qb, DH, 32);

  v8f oacc[4] = {};
  float m_i[8], l_i[8];
#pragma unroll
  for (int i = 0; i < 8; ++i) {
    m_i[i] = -1e30f;
    l_i[i] = 0.0f;
  }

  // All 4 waves iterate to the block's max key range; masking handles edges.
  int niter = (HIST + qt4 * 64 + 64 + 31) >> 5;

#if USE_TDM
  if (wave == 0) {  // prologue: DMA first K/V block into buffer 0
    tdm_load_2d(0u, kb, DH, 32, DH);
    tdm_load_2d(8192u, vb, 32, DH, SEQ_LEN);
  }
#endif

  for (int it = 0; it < niter; ++it) {
    int j0 = it << 5;
#if USE_TDM
    int cur = it & 1;
    if (wave == 0) {
      if (it + 1 < niter) {  // overlap next DMA with this block's WMMAs
        int nb = (it + 1) & 1;
        tdm_load_2d((unsigned)(nb * 4096), kb + (size_t)(j0 + 32) * DH, DH, 32,
                    DH);
        tdm_load_2d((unsigned)(8192 + nb * 4096), vb + (j0 + 32), 32, DH,
                    SEQ_LEN);
        __builtin_amdgcn_s_wait_tensorcnt(2);  // current block complete
      } else {
        __builtin_amdgcn_s_wait_tensorcnt(0);
      }
    }
#else
    int cur = 0;
    {  // cooperative fallback staging (prev trailing barrier protects buffers)
      int t = threadIdx.x;
      for (int c = t; c < 256; c += 128) {  // K: 32 rows x 64 halves
        int r = c >> 3, off = (c & 7) * 8;
        const half_t* src = kb + (size_t)(j0 + r) * DH + off;
        _Float16* dst = smem + r * 64 + off;
#pragma unroll
        for (int u = 0; u < 8; ++u) dst[u] = src[u];
      }
      for (int c = t; c < 256; c += 128) {  // V: 64 dim-rows x 32 halves
        int d = c >> 2, off = (c & 3) * 8;
        const half_t* src = vb + (size_t)d * SEQ_LEN + j0 + off;
        _Float16* dst = smem + 4096 + d * 32 + off;
#pragma unroll
        for (int u = 0; u < 8; ++u) dst[u] = src[u];
      }
    }
#endif
    __syncthreads();  // K/V block visible to all waves

    const _Float16* kB = smem + cur * 2048;
    const _Float16* vB = smem + 4096 + cur * 2048;
    _Float16* pw = smem + 8192 + wave * 512;

    v16h bk00 = frag_from_f16(kB, 64, 0);
    v16h bk01 = frag_from_f16(kB, 64, 32);
    v16h bk10 = frag_from_f16(kB + 16 * 64, 64, 0);
    v16h bk11 = frag_from_f16(kB + 16 * 64, 64, 32);

    v8f s0 = {};
    v8f s1 = {};
    s0 = wmma_f16(qa0, bk00, s0);
    s0 = wmma_f16(qa1, bk01, s0);
    s1 = wmma_f16(qa0, bk10, s1);
    s1 = wmma_f16(qa1, bk11, s1);

    float alpha[8];
#pragma unroll
    for (int i = 0; i < 8; ++i) {
      int row = i0 + rh + i;  // query index
      float a0 = (j0 + col <= HIST + row) ? s0[i] : -1e30f;
      float a1 = (j0 + 16 + col <= HIST + row) ? s1[i] : -1e30f;
      float mx = fmaxf(a0, a1);
#pragma unroll
      for (int m = 1; m < 16; m <<= 1) mx = fmaxf(mx, __shfl_xor(mx, m, 32));
      float mn = fmaxf(m_i[i], mx);
      float al = __expf(m_i[i] - mn);
      float p0 = __expf(a0 - mn);
      float p1 = __expf(a1 - mn);
      float rs = p0 + p1;
#pragma unroll
      for (int m = 1; m < 16; m <<= 1) rs += __shfl_xor(rs, m, 32);
      m_i[i] = mn;
      l_i[i] = l_i[i] * al + rs;
      alpha[i] = al;
      pw[(rh + i) * 32 + col] = (_Float16)p0;
      pw[(rh + i) * 32 + 16 + col] = (_Float16)p1;
    }
#pragma unroll
    for (int f = 0; f < 4; ++f)
#pragma unroll
      for (int i = 0; i < 8; ++i) oacc[f][i] *= alpha[i];

    // Re-read P in A-fragment layout (wave-local; LDS is in-order per wave).
    v16h pa;
    {
      int r = lane & 15;
      int cb = (lane & 16) ? 8 : 0;
      const _Float16* pp = pw + r * 32 + cb;
#pragma unroll
      for (int i = 0; i < 8; ++i) pa[i] = pp[i];
#pragma unroll
      for (int i = 0; i < 8; ++i) pa[8 + i] = pp[16 + i];
    }
#pragma unroll
    for (int f = 0; f < 4; ++f) {
      v16h vf = frag_from_f16(vB + f * 16 * 32, 32, 0);
      oacc[f] = wmma_f16(pa, vf, oacc[f]);
    }
    __syncthreads();  // all waves done with this buffer before reuse
  }

  int b_ = bh >> 4, h = bh & 15;
#pragma unroll
  for (int i = 0; i < 8; ++i) {
    float invl = 1.0f / l_i[i];
    int row = i0 + rh + i;
#pragma unroll
    for (int f = 0; f < 4; ++f) {
      int d = f * 16 + col;
      o[((size_t)b_ * LATENT + row) * DIM + h * DH + d] =
          (half_t)(oacc[f][i] * invl);
    }
  }
}

// Y[2048,1024] = O(f16) @ Wo^T(f16) + bo, fp32 out.
__global__ __launch_bounds__(32) void out_proj_kernel(
    const half_t* __restrict__ A, const half_t* __restrict__ Wo16,
    const float* __restrict__ bo, float* __restrict__ Y) {
  int nt = blockIdx.x & 15;
  int mt = blockIdx.x >> 4;
  int lane = threadIdx.x & 31;
  int col = lane & 15;
  int rh = (lane & 16) ? 8 : 0;
  const half_t* Ab = A + (size_t)(mt * 16) * DIM;
  const half_t* Bb = Wo16 + (size_t)(nt * 64) * DIM;
  v8f acc[4] = {};
  for (int k = 0; k < DIM; k += 32) {
    v16h a = frag_from_f16(Ab, DIM, k);
#pragma unroll
    for (int f = 0; f < 4; ++f) {
      v16h bf = frag_from_f16(Bb + (size_t)(f * 16) * DIM, DIM, k);
      acc[f] = wmma_f16(a, bf, acc[f]);
    }
  }
#pragma unroll
  for (int i = 0; i < 8; ++i) {
    int row = mt * 16 + rh + i;
#pragma unroll
    for (int f = 0; f < 4; ++f) {
      int e = nt * 64 + f * 16 + col;
      Y[(size_t)row * DIM + e] = acc[f][i] + bo[e];
    }
  }
}

extern "C" void kernel_launch(void* const* d_in, const int* in_sizes, int n_in,
                              void* d_out, int out_size, void* d_ws,
                              size_t ws_size, hipStream_t stream) {
  const float* x = (const float*)d_in[0];
  const float* Wq = (const float*)d_in[1];
  const float* Wkv = (const float*)d_in[2];
  const float* Wo = (const float*)d_in[3];
  const float* bo = (const float*)d_in[4];
  const float* gamma = (const float*)d_in[5];
  const float* beta = (const float*)d_in[6];
  float* y = (float*)d_out;

  // Workspace layout (f16 elements):
  // q[2,16,1024,64] | kvn[2,16,4096,64] | kvnT[2,16,64,4096] | o[2,1024,1024]
  // | x16[2,4096,1024] | wq16 | wkv16 | wo16       (~60 MB total)
  half_t* q_ws = (half_t*)d_ws;
  half_t* kvn_ws = q_ws + (size_t)BATCH * HEADS * LATENT * DH;
  half_t* kvnT_ws = kvn_ws + (size_t)BATCH * HEADS * SEQ_LEN * DH;
  half_t* o_ws = kvnT_ws + (size_t)BATCH * HEADS * SEQ_LEN * DH;
  half_t* x16 = o_ws + (size_t)BATCH * LATENT * DIM;
  half_t* wq16 = x16 + (size_t)BATCH * SEQ_LEN * DIM;
  half_t* wkv16 = wq16 + (size_t)DIM * DIM;
  half_t* wo16 = wkv16 + (size_t)DIM * DIM;

  int nx = BATCH * SEQ_LEN * DIM;
  int nw = DIM * DIM;
  cvt_kernel<<<(nx / 8 + 255) / 256, 256, 0, stream>>>(x, x16, nx);
  cvt_kernel<<<(nw / 8 + 255) / 256, 256, 0, stream>>>(Wq, wq16, nw);
  cvt_kernel<<<(nw / 8 + 255) / 256, 256, 0, stream>>>(Wkv, wkv16, nw);
  cvt_kernel<<<(nw / 8 + 255) / 256, 256, 0, stream>>>(Wo, wo16, nw);

  // kv projection + RoPE + LayerNorm over all 4096 positions
  proj_kernel<<<(BATCH * SEQ_LEN / 16) * HEADS, 32, 0, stream>>>(
      x16, wkv16, gamma, beta, kvn_ws, kvnT_ws, SEQ_LEN, 0, 1, 1.0f);
  // q projection + RoPE + scale over last 1024 positions only
  proj_kernel<<<(BATCH * LATENT / 16) * HEADS, 32, 0, stream>>>(
      x16, wq16, gamma, beta, q_ws, nullptr, LATENT, HIST, 0, 0.125f);
  // flash attention (4 q-tiles per block share LDS-staged K/V)
  attn_kernel<<<BATCH * HEADS * (LATENT / 64), 128, 0, stream>>>(
      q_ws, kvn_ws, kvnT_ws, o_ws);
  // output projection + bias
  out_proj_kernel<<<(BATCH * LATENT / 16) * (DIM / 64), 32, 0, stream>>>(
      o_ws, wo16, bo, y);
}
